// SparseMixtureOfExperts_9929964388698
// MI455X (gfx1250) — matrix operand order repaired and checked by reading, not compiled, not run on hardware
//
#include <hip/hip_runtime.h>
#include <hip/hip_bf16.h>

#define N_TOK   16384
#define D_IN    1024
#define HID     256
#define D_OUT   1024
#define NEXP    8
#define TILE_M  32
#define MAX_TILES (N_TOK / TILE_M)   // 512 tiles per expert worst case

typedef __attribute__((ext_vector_type(16))) __bf16 v16bf;
typedef __attribute__((ext_vector_type(8)))  __bf16 v8bf;
typedef __attribute__((ext_vector_type(8)))  float  v8f;

// ---------------------------------------------------------------- zero out
__global__ __launch_bounds__(256) void k_zero4(float4* __restrict__ p, int n4) {
    int i = blockIdx.x * 256 + threadIdx.x;
    if (i < n4) p[i] = make_float4(0.f, 0.f, 0.f, 0.f);
}

// ------------------------------------------- transpose+convert weights to bf16
// W1 [E][K=1024][H=256]  -> W1t [E][H][K]   (K contiguous per output column)
// W2 [E][K=256][O=1024]  -> W2t [E][O][K]
__global__ __launch_bounds__(256) void k_wconv(const float* __restrict__ W1,
                                               const float* __restrict__ W2,
                                               __bf16* __restrict__ W1t,
                                               __bf16* __restrict__ W2t,
                                               int* __restrict__ counts) {
    int tid = blockIdx.x * 256 + threadIdx.x;
    if (blockIdx.x == 0 && threadIdx.x < NEXP) counts[threadIdx.x] = 0;
    const int n1 = NEXP * D_IN * HID;            // 2M
    const int n2 = NEXP * HID * D_OUT;           // 2M
    if (tid < n1) {
        int e = tid / (HID * D_IN);
        int r = tid % (HID * D_IN);
        int n = r / D_IN, k = r % D_IN;
        W1t[tid] = (__bf16)W1[(size_t)e * D_IN * HID + (size_t)k * HID + n];
    } else if (tid < n1 + n2) {
        int t = tid - n1;
        int e = t / (D_OUT * HID);
        int r = t % (D_OUT * HID);
        int o = r / HID, k = r % HID;
        W2t[t] = (__bf16)W2[(size_t)e * HID * D_OUT + (size_t)k * D_OUT + o];
    }
}

// ---------------------------------------------------------------- gating
// One wave32 per token: logits, top-2 softmax, expert dispatch lists,
// and x f32 -> bf16 conversion for the WMMA phase.
__global__ __launch_bounds__(256) void k_gate(const float* __restrict__ x,
                                              const float* __restrict__ Wg,
                                              const float* __restrict__ bg,
                                              __bf16* __restrict__ xb,
                                              float* __restrict__ gate_logits,
                                              int*   __restrict__ counts,
                                              int*   __restrict__ tok_ids,
                                              float* __restrict__ gate_w) {
    const int tok  = (blockIdx.x * 256 + threadIdx.x) >> 5;
    const int lane = threadIdx.x & 31;
    if (tok >= N_TOK) return;
    const float* xr  = x  + (size_t)tok * D_IN;
    __bf16*      xbr = xb + (size_t)tok * D_IN;

    float acc[NEXP];
#pragma unroll
    for (int e = 0; e < NEXP; e++) acc[e] = 0.f;
#pragma unroll 4
    for (int j = 0; j < D_IN / 32; j++) {
        const int d = lane + j * 32;
        const float v = xr[d];
        xbr[d] = (__bf16)v;
        const float* wg = Wg + d * NEXP;
#pragma unroll
        for (int e = 0; e < NEXP; e++) acc[e] += v * wg[e];
    }
#pragma unroll
    for (int e = 0; e < NEXP; e++) {
        float a = acc[e];
#pragma unroll
        for (int sh = 16; sh > 0; sh >>= 1) a += __shfl_xor(a, sh, 32);
        acc[e] = a + bg[e];
    }
    if (lane == 0) {
        float* gl = gate_logits + (size_t)tok * NEXP;
#pragma unroll
        for (int e = 0; e < NEXP; e++) gl[e] = acc[e];
        float b1v = -3.4e38f; int i1 = 0;
#pragma unroll
        for (int e = 0; e < NEXP; e++) if (acc[e] > b1v) { b1v = acc[e]; i1 = e; }
        float b2v = -3.4e38f; int i2 = 0;
#pragma unroll
        for (int e = 0; e < NEXP; e++) if (e != i1 && acc[e] > b2v) { b2v = acc[e]; i2 = e; }
        const float a2 = expf(b2v - b1v);           // vals sorted desc -> stable
        const float w1 = 1.f / (1.f + a2);
        const float w2 = a2  / (1.f + a2);
        int s1 = atomicAdd(&counts[i1], 1);
        tok_ids[i1 * N_TOK + s1] = tok;  gate_w[i1 * N_TOK + s1] = w1;
        int s2 = atomicAdd(&counts[i2], 1);
        tok_ids[i2 * N_TOK + s2] = tok;  gate_w[i2 * N_TOK + s2] = w2;
    }
}

// ------------------------------------------------------------- expert GEMMs
// Block = (expert e, 32-token tile). 8 waves.
//   GEMM1: h[32,256] = relu(A[32,1024] * W1t + b1)   (bf16 WMMA, f32 acc)
//   GEMM2: y[32,1024] = h * W2t + b2 ; atomic scatter g*y into out
__global__ __launch_bounds__(256) void k_expert(const __bf16* __restrict__ xb,
                                                const __bf16* __restrict__ W1t,
                                                const __bf16* __restrict__ W2t,
                                                const float*  __restrict__ b1,
                                                const float*  __restrict__ b2,
                                                const int*    __restrict__ counts,
                                                const int*    __restrict__ tok_ids,
                                                const float*  __restrict__ gate_w,
                                                float* __restrict__ out) {
    __shared__ int    s_tok[TILE_M];
    __shared__ float  s_gate[TILE_M];
    __shared__ __bf16 s_h[TILE_M][HID];          // 16 KB

    const int e    = blockIdx.x / MAX_TILES;
    const int tile = blockIdx.x % MAX_TILES;
    const int cnt  = counts[e];
    if (tile * TILE_M >= cnt) return;            // uniform early exit
    const int rem = cnt - tile * TILE_M;

    const int tidx = threadIdx.x;
    const int wave = tidx >> 5;
    const int lane = tidx & 31;
    const int lh   = lane >> 4;                  // lane half (K split)
    const int ln   = lane & 15;                  // row (A) / col (B,C,D)
    const int mblk = wave & 1;                   // which 16-row block

    if (tidx < TILE_M) {
        const int slot = tile * TILE_M + tidx;
        const bool v = tidx < rem;
        s_tok[tidx]  = v ? tok_ids[e * N_TOK + slot] : 0;
        s_gate[tidx] = v ? gate_w[e * N_TOK + slot]  : 0.f;
    }
    __syncthreads();

    // ---------------- GEMM1 -> s_h -------------------------------------
    {
        const int row = mblk * 16 + ln;
        const __bf16* arow = xb + (size_t)s_tok[row] * D_IN;
        const int ng = wave >> 1;                // 4 waves share each m-block
        v8f acc[4] = {};
        for (int k0 = 0; k0 < D_IN; k0 += 32) {
            // A fragment (16x32 bf16): lane<16 holds K {k0..+15}, lane>=16 K {k0+16..+31}
            union { v16bf v; struct { v8bf lo; v8bf hi; } p; } a;
            a.p.lo = *(const v8bf*)(arow + k0 + (lh << 4));
            a.p.hi = *(const v8bf*)(arow + k0 + 8 + (lh << 4));
#pragma unroll
            for (int i = 0; i < 4; i++) {
                const int col = (ng * 4 + i) * 16 + ln;
                const __bf16* bcol = W1t + ((size_t)e * HID + col) * D_IN;
                // B fragment (32x16): lanes 0-15 K=k0..+15, lanes 16-31 K+16..+31
                v16bf b = *(const v16bf*)(bcol + k0 + (lh << 4));
                acc[i] = __builtin_amdgcn_wmma_f32_16x16x32_bf16(
                    false, a.v, false, b, (short)0, acc[i], false, false);
            }
        }
#pragma unroll
        for (int i = 0; i < 4; i++) {
            const int col  = (ng * 4 + i) * 16 + ln;
            const float bv = b1[e * HID + col];
#pragma unroll
            for (int r = 0; r < 8; r++) {        // C/D: vgpr r -> M = r + 8*lh
                const int m = mblk * 16 + (lh << 3) + r;
                float h = acc[i][r] + bv;
                s_h[m][col] = (__bf16)(h > 0.f ? h : 0.f);
            }
        }
    }
    __syncthreads();

    // ---------------- GEMM2 + scatter ----------------------------------
    {
        union AF { v16bf v; struct { v8bf lo; v8bf hi; } p; };
        AF a[8];                                  // hoist all K of h (256)
        const int row = mblk * 16 + ln;
#pragma unroll
        for (int ks = 0; ks < 8; ks++) {
            const int k0 = ks * 32;
            a[ks].p.lo = *(const v8bf*)(&s_h[row][k0 + (lh << 4)]);
            a[ks].p.hi = *(const v8bf*)(&s_h[row][k0 + 8 + (lh << 4)]);
        }
        int   ltok[8]; float lg[8]; bool lv[8];
#pragma unroll
        for (int r = 0; r < 8; r++) {
            const int m = mblk * 16 + (lh << 3) + r;
            ltok[r] = s_tok[m]; lg[r] = s_gate[m]; lv[r] = (m < rem);
        }
        const int ngrp = wave >> 1;               // 16 n-tiles per wave
#pragma unroll 1
        for (int j = 0; j < 16; j++) {
            const int o = (ngrp * 16 + j) * 16 + ln;
            const __bf16* bcol = W2t + ((size_t)e * D_OUT + o) * HID;
            v8f acc = {};
#pragma unroll
            for (int ks = 0; ks < 8; ks++) {
                v16bf b = *(const v16bf*)(bcol + ks * 32 + (lh << 4));
                acc = __builtin_amdgcn_wmma_f32_16x16x32_bf16(
                    false, a[ks].v, false, b, (short)0, acc, false, false);
            }
            const float bv = b2[e * D_OUT + o];
#pragma unroll
            for (int r = 0; r < 8; r++) {
                if (lv[r])
                    atomicAdd(&out[(size_t)ltok[r] * D_OUT + o], lg[r] * (acc[r] + bv));
            }
        }
    }
}

// ---------------------------------------------------------------- launcher
extern "C" void kernel_launch(void* const* d_in, const int* in_sizes, int n_in,
                              void* d_out, int out_size, void* d_ws, size_t ws_size,
                              hipStream_t stream) {
    const float* x  = (const float*)d_in[0];
    const float* Wg = (const float*)d_in[1];
    const float* bg = (const float*)d_in[2];
    const float* W1 = (const float*)d_in[3];
    const float* b1 = (const float*)d_in[4];
    const float* W2 = (const float*)d_in[5];
    const float* b2 = (const float*)d_in[6];

    float* out = (float*)d_out;
    float* gl  = out + (size_t)N_TOK * D_OUT;     // gate_logits tail

    char* ws = (char*)d_ws;
    size_t off = 0;
    __bf16* xb  = (__bf16*)(ws + off); off += (size_t)N_TOK * D_IN * 2;       // 32 MB
    __bf16* W1t = (__bf16*)(ws + off); off += (size_t)NEXP * D_IN * HID * 2;  // 4 MB
    __bf16* W2t = (__bf16*)(ws + off); off += (size_t)NEXP * HID * D_OUT * 2; // 4 MB
    int*   counts = (int*)(ws + off);  off += 256;
    int*   tok    = (int*)(ws + off);  off += (size_t)NEXP * N_TOK * 4;       // 512 KB
    float* gw     = (float*)(ws + off);

    const int n4 = N_TOK * D_OUT / 4;
    k_zero4<<<(n4 + 255) / 256, 256, 0, stream>>>((float4*)out, n4);

    const int nconv = NEXP * D_IN * HID + NEXP * HID * D_OUT;
    k_wconv<<<(nconv + 255) / 256, 256, 0, stream>>>(W1, W2, W1t, W2t, counts);

    k_gate<<<(N_TOK * 32) / 256, 256, 0, stream>>>(x, Wg, bg, xb, gl, counts, tok, gw);

    k_expert<<<NEXP * MAX_TILES, 256, 0, stream>>>(xb, W1t, W2t, b1, b2,
                                                   counts, tok, gw, out);
}
